// DOMTransformerLayer_8323646619772
// MI455X (gfx1250) — matrix operand order repaired and checked by reading, not compiled, not run on hardware
//
#include <hip/hip_runtime.h>
#include <hip/hip_bf16.h>

// ---------------- types ----------------
typedef _Float16 v16h __attribute__((ext_vector_type(16)));
typedef _Float16 v8h  __attribute__((ext_vector_type(8)));
typedef float    v8f  __attribute__((ext_vector_type(8)));

#define D_MODEL 1024
#define N_HEADS 16
#define HEAD_DIM 64
#define SEQ 2048
#define BATCH 4
#define D_FF 4096
#define M_ROWS (BATCH * SEQ)   // 8192
#define LDT 40                 // LDS row stride in halves (80B, 16B-aligned)

__device__ __forceinline__ v8f zero8() {
    v8f z = {0.f, 0.f, 0.f, 0.f, 0.f, 0.f, 0.f, 0.f};
    return z;
}

__device__ __forceinline__ v8f wmma_f16f32(v16h a, v16h b, v8f c) {
    return __builtin_amdgcn_wmma_f32_16x16x32_f16(false, a, false, b, (short)0, c,
                                                  false, false);
}

// A fragment (16x32): lane m = l&15, 16 contiguous K halves at K = hi*16
__device__ __forceinline__ v16h load_a16(const _Float16* p) {
    v8h lo = *reinterpret_cast<const v8h*>(p);
    v8h hi = *reinterpret_cast<const v8h*>(p + 8);
    v16h r;
#pragma unroll
    for (int i = 0; i < 8; ++i) { r[i] = lo[i]; r[i + 8] = hi[i]; }
    return r;
}
// B fragment (32x16): lane n = l&15, 16 contiguous K halves at K = hi*16
__device__ __forceinline__ v16h load_b16(const _Float16* p) {
    return *reinterpret_cast<const v16h*>(p);
}

// ---------------- weight convert + transpose: Wt[n*K+k] = (f16) W[k*N+n] ----------
__global__ __launch_bounds__(256) void wtrans_kernel(const float* __restrict__ W,
                                                     _Float16* __restrict__ Wt,
                                                     int K, int N) {
    __shared__ float tile[32][33];
    const int tx = threadIdx.x & 31, ty = threadIdx.x >> 5;
    const int k0 = blockIdx.x * 32, n0 = blockIdx.y * 32;
#pragma unroll
    for (int j = 0; j < 4; ++j)
        tile[ty + 8 * j][tx] = W[(size_t)(k0 + ty + 8 * j) * N + n0 + tx];
    __syncthreads();
#pragma unroll
    for (int j = 0; j < 4; ++j)
        Wt[(size_t)(n0 + ty + 8 * j) * K + k0 + tx] = (_Float16)tile[tx][ty + 8 * j];
}

// ---------------- f32 -> f16 elementwise ----------------
__global__ __launch_bounds__(256) void cvt16_kernel(const float* __restrict__ in,
                                                    _Float16* __restrict__ out,
                                                    int n4) {
    const int i = blockIdx.x * 256 + threadIdx.x;
    if (i < n4) {
        const float4 v = reinterpret_cast<const float4*>(in)[i];
        _Float16* o = out + (size_t)i * 4;
        o[0] = (_Float16)v.x; o[1] = (_Float16)v.y;
        o[2] = (_Float16)v.z; o[3] = (_Float16)v.w;
    }
}

// ---------------- WMMA GEMM ----------------
// C[M,N] = A[M,K] @ Wt[N,K]^T + bias.  A,Wt f16 row-major. Block 128x128, BK=32,
// 256 thr = 8 waves, double-buffered LDS.
// MODE 0: f32 C out.  MODE 1: f16 gelu(C) out.  MODE 2: f16 q/k/v_t scatter.
template <int MODE>
__global__ __launch_bounds__(256) void gemm_wmma_kernel(
    const _Float16* __restrict__ A, const _Float16* __restrict__ Wt,
    const float* __restrict__ bias, float* __restrict__ Cout,
    _Float16* __restrict__ H16, _Float16* __restrict__ qh,
    _Float16* __restrict__ kh, _Float16* __restrict__ vt, int M, int N, int K) {
    __shared__ __align__(16) _Float16 As[2][128 * LDT];
    __shared__ __align__(16) _Float16 Bs[2][128 * LDT];

    const int t = threadIdx.x;
    const int bn = blockIdx.x * 128;
    const int bm = blockIdx.y * 128;
    const int wave = t >> 5, lane = t & 31, lanei = lane & 15, hi = lane >> 4;
    const int wm = (wave >> 2) * 64;
    const int wn = (wave & 3) * 32;

    // staging unit: thread t covers units 2t, 2t+1; each unit = 8 halves (16B)
    const int u0 = t * 2;
    const int srow0 = u0 >> 2, sc0 = (u0 & 3) * 8;
    const int srow1 = (u0 + 1) >> 2, sc1 = ((u0 + 1) & 3) * 8;
    const _Float16* gA0 = A + (size_t)(bm + srow0) * K + sc0;
    const _Float16* gA1 = A + (size_t)(bm + srow1) * K + sc1;
    const _Float16* gB0 = Wt + (size_t)(bn + srow0) * K + sc0;
    const _Float16* gB1 = Wt + (size_t)(bn + srow1) * K + sc1;

    v8f acc[4][2];
#pragma unroll
    for (int mt = 0; mt < 4; ++mt)
#pragma unroll
        for (int nt = 0; nt < 2; ++nt) acc[mt][nt] = zero8();

    const int KT = K >> 5;
    v8h ra0, ra1, rb0, rb1;
    // prologue: tile 0 -> regs -> buf 0
    ra0 = *reinterpret_cast<const v8h*>(gA0);
    ra1 = *reinterpret_cast<const v8h*>(gA1);
    rb0 = *reinterpret_cast<const v8h*>(gB0);
    rb1 = *reinterpret_cast<const v8h*>(gB1);
    *reinterpret_cast<v8h*>(&As[0][srow0 * LDT + sc0]) = ra0;
    *reinterpret_cast<v8h*>(&As[0][srow1 * LDT + sc1]) = ra1;
    *reinterpret_cast<v8h*>(&Bs[0][srow0 * LDT + sc0]) = rb0;
    *reinterpret_cast<v8h*>(&Bs[0][srow1 * LDT + sc1]) = rb1;

    for (int kt = 0; kt < KT; ++kt) {
        __syncthreads();
        const int cur = kt & 1;
        if (kt + 1 < KT) {
            const int ko = (kt + 1) * 32;
            ra0 = *reinterpret_cast<const v8h*>(gA0 + ko);
            ra1 = *reinterpret_cast<const v8h*>(gA1 + ko);
            rb0 = *reinterpret_cast<const v8h*>(gB0 + ko);
            rb1 = *reinterpret_cast<const v8h*>(gB1 + ko);
            // warm L2/WGP$ for the tile after next (speculative, fault-free)
            __builtin_prefetch(gA0 + ko + 32, 0, 1);
            __builtin_prefetch(gB0 + ko + 32, 0, 1);
        }
        const _Float16* as = As[cur];
        const _Float16* bs = Bs[cur];
        v16h af[4], bf[2];
#pragma unroll
        for (int mt = 0; mt < 4; ++mt)
            af[mt] = load_a16(&as[(wm + mt * 16 + lanei) * LDT + hi * 16]);
#pragma unroll
        for (int nt = 0; nt < 2; ++nt)
            bf[nt] = load_b16(&bs[(wn + nt * 16 + lanei) * LDT + hi * 16]);
#pragma unroll
        for (int mt = 0; mt < 4; ++mt)
#pragma unroll
            for (int nt = 0; nt < 2; ++nt)
                acc[mt][nt] = wmma_f16f32(af[mt], bf[nt], acc[mt][nt]);
        if (kt + 1 < KT) {
            const int nxt = cur ^ 1;
            *reinterpret_cast<v8h*>(&As[nxt][srow0 * LDT + sc0]) = ra0;
            *reinterpret_cast<v8h*>(&As[nxt][srow1 * LDT + sc1]) = ra1;
            *reinterpret_cast<v8h*>(&Bs[nxt][srow0 * LDT + sc0]) = rb0;
            *reinterpret_cast<v8h*>(&Bs[nxt][srow1 * LDT + sc1]) = rb1;
        }
    }

    // epilogue: C layout -> row m = r + 8*hi, col n = lanei
#pragma unroll
    for (int mt = 0; mt < 4; ++mt) {
#pragma unroll
        for (int nt = 0; nt < 2; ++nt) {
            const int colg = bn + wn + nt * 16 + lanei;
            const float bv = bias[colg];
#pragma unroll
            for (int r = 0; r < 8; ++r) {
                const int rowg = bm + wm + mt * 16 + r + hi * 8;
                float val = acc[mt][nt][r] + bv;
                if (MODE == 0) {
                    Cout[(size_t)rowg * N + colg] = val;
                } else if (MODE == 1) {
                    H16[(size_t)rowg * N + colg] =
                        (_Float16)(0.5f * val * (1.0f + erff(val * 0.70710678f)));
                } else {
                    const int b = rowg >> 11, s = rowg & (SEQ - 1);
                    if (colg < D_MODEL) {
                        int hh = colg >> 6, dd = colg & 63;
                        qh[((size_t)(b * N_HEADS + hh) * SEQ + s) * HEAD_DIM + dd] =
                            (_Float16)val;
                    } else if (colg < 2 * D_MODEL) {
                        int c2 = colg - D_MODEL, hh = c2 >> 6, dd = c2 & 63;
                        kh[((size_t)(b * N_HEADS + hh) * SEQ + s) * HEAD_DIM + dd] =
                            (_Float16)val;
                    } else {
                        int c2 = colg - 2 * D_MODEL, hh = c2 >> 6, dd = c2 & 63;
                        vt[((size_t)(b * N_HEADS + hh) * HEAD_DIM + dd) * SEQ + s] =
                            (_Float16)val;
                    }
                }
            }
        }
    }
}

// ---------------- masked flash attention (segment-band sparse) ----------------
// segment_ids are sorted along s => segments are contiguous runs. A k-tile whose
// segment range doesn't intersect the q-tile's range is fully masked: skip it.
__global__ __launch_bounds__(256) void attn_kernel(
    const _Float16* __restrict__ qh, const _Float16* __restrict__ kh,
    const _Float16* __restrict__ vt, const int* __restrict__ seg,
    _Float16* __restrict__ o) {
    __shared__ __align__(16) _Float16 Ps[8][16 * 32];

    const int t = threadIdx.x;
    const int w = t >> 5, lane = t & 31, lanei = lane & 15, hi = lane >> 4;
    const int bh = blockIdx.y, b = bh >> 4, h = bh & 15;
    const int qr = blockIdx.x * 128 + w * 16;

    const _Float16* qb = qh + ((size_t)bh * SEQ + qr) * HEAD_DIM;
    const v16h aQ0 = load_a16(qb + lanei * HEAD_DIM + hi * 16);
    const v16h aQ1 = load_a16(qb + lanei * HEAD_DIM + 32 + hi * 16);

    int segq[8];
#pragma unroll
    for (int r = 0; r < 8; ++r) segq[r] = seg[b * SEQ + qr + r + hi * 8];

    // q-tile segment range (ids sorted: row 0 has min of this half, row 7 max)
    int sqmin = segq[0], sqmax = segq[0];
#pragma unroll
    for (int r = 1; r < 8; ++r) {
        sqmin = min(sqmin, segq[r]);
        sqmax = max(sqmax, segq[r]);
    }
    sqmin = min(sqmin, __shfl_xor(sqmin, 16, 32));  // combine the two half-waves
    sqmax = max(sqmax, __shfl_xor(sqmax, 16, 32));

    float m_i[8], l_i[8];
    v8f accO[4];
#pragma unroll
    for (int r = 0; r < 8; ++r) { m_i[r] = -1e30f; l_i[r] = 0.f; }
#pragma unroll
    for (int c4 = 0; c4 < 4; ++c4) accO[c4] = zero8();

    const _Float16* kb = kh + (size_t)bh * SEQ * HEAD_DIM;
    const _Float16* vb = vt + (size_t)bh * HEAD_DIM * SEQ;
    _Float16* myP = Ps[w];
    const float scale = 0.125f;

    for (int kt = 0; kt < SEQ / 32; ++kt) {
        const int kbase = kt * 32;
        // wave-uniform band test: sorted ids => tile range = [seg[kbase], seg[kbase+31]]
        const int ks_lo = seg[b * SEQ + kbase];
        const int ks_hi = seg[b * SEQ + kbase + 31];
        if (ks_hi < sqmin || ks_lo > sqmax) continue;  // fully masked tile

        float sc0[8], sc1[8];
        {
            const int key = kbase + lanei;
            const _Float16* kp = kb + (size_t)key * HEAD_DIM + hi * 16;
            v8f c = zero8();
            c = wmma_f16f32(aQ0, load_b16(kp), c);
            c = wmma_f16f32(aQ1, load_b16(kp + 32), c);
            const int sk = seg[b * SEQ + key];
#pragma unroll
            for (int r = 0; r < 8; ++r)
                sc0[r] = (segq[r] == sk) ? c[r] * scale : -1e30f;
        }
        {
            const int key = kbase + 16 + lanei;
            const _Float16* kp = kb + (size_t)key * HEAD_DIM + hi * 16;
            v8f c = zero8();
            c = wmma_f16f32(aQ0, load_b16(kp), c);
            c = wmma_f16f32(aQ1, load_b16(kp + 32), c);
            const int sk = seg[b * SEQ + key];
#pragma unroll
            for (int r = 0; r < 8; ++r)
                sc1[r] = (segq[r] == sk) ? c[r] * scale : -1e30f;
        }
        float p0a[8], p1a[8];
#pragma unroll
        for (int r = 0; r < 8; ++r) {
            float tm = fmaxf(sc0[r], sc1[r]);
            tm = fmaxf(tm, __shfl_xor(tm, 1, 32));
            tm = fmaxf(tm, __shfl_xor(tm, 2, 32));
            tm = fmaxf(tm, __shfl_xor(tm, 4, 32));
            tm = fmaxf(tm, __shfl_xor(tm, 8, 32));
            const float mn = fmaxf(m_i[r], tm);
            const float sf = __expf(m_i[r] - mn);
            const float p0 = __expf(sc0[r] - mn);
            const float p1 = __expf(sc1[r] - mn);
            float rs = p0 + p1;
            rs += __shfl_xor(rs, 1, 32);
            rs += __shfl_xor(rs, 2, 32);
            rs += __shfl_xor(rs, 4, 32);
            rs += __shfl_xor(rs, 8, 32);
            l_i[r] = l_i[r] * sf + rs;
            m_i[r] = mn;
            accO[0][r] *= sf; accO[1][r] *= sf;
            accO[2][r] *= sf; accO[3][r] *= sf;
            p0a[r] = p0; p1a[r] = p1;
        }
#pragma unroll
        for (int r = 0; r < 8; ++r) {
            myP[(r + hi * 8) * 32 + lanei]      = (_Float16)p0a[r];
            myP[(r + hi * 8) * 32 + 16 + lanei] = (_Float16)p1a[r];
        }
        asm volatile("s_wait_dscnt 0" ::: "memory");
        const v16h pA = load_a16(myP + lanei * 32 + hi * 16);
#pragma unroll
        for (int c4 = 0; c4 < 4; ++c4) {
            const _Float16* vp =
                vb + (size_t)(c4 * 16 + lanei) * SEQ + kbase + hi * 16;
            accO[c4] = wmma_f16f32(pA, load_b16(vp), accO[c4]);
        }
    }

#pragma unroll
    for (int r = 0; r < 8; ++r) {
        const float inv = (l_i[r] > 0.f) ? (1.0f / l_i[r]) : 0.f;
        const size_t row = (size_t)b * SEQ + qr + r + hi * 8;
#pragma unroll
        for (int c4 = 0; c4 < 4; ++c4)
            o[row * D_MODEL + h * HEAD_DIM + c4 * 16 + lanei] =
                (_Float16)(accO[c4][r] * inv);
    }
}

// ---------------- fused residual + LayerNorm (f32 out + optional f16 out) -------
__global__ __launch_bounds__(256) void ln_kernel(
    const float* __restrict__ xa, const float* __restrict__ xb,
    const float* __restrict__ g, const float* __restrict__ be,
    float* __restrict__ out, _Float16* __restrict__ outh) {
    const int row = blockIdx.x, t = threadIdx.x;
    const float* pa = xa + (size_t)row * D_MODEL;
    const float* pb = xb + (size_t)row * D_MODEL;
    float v[4], s = 0.f, q = 0.f;
#pragma unroll
    for (int i = 0; i < 4; ++i) {
        const int idx = t + i * 256;
        const float x = pa[idx] + pb[idx];
        v[i] = x; s += x; q += x * x;
    }
#pragma unroll
    for (int off = 16; off >= 1; off >>= 1) {
        s += __shfl_xor(s, off, 32);
        q += __shfl_xor(q, off, 32);
    }
    __shared__ float sh[16];
    const int w = t >> 5, lane = t & 31;
    if (lane == 0) { sh[w] = s; sh[8 + w] = q; }
    __syncthreads();
    float ts = 0.f, tq = 0.f;
#pragma unroll
    for (int i = 0; i < 8; ++i) { ts += sh[i]; tq += sh[8 + i]; }
    const float mu = ts * (1.0f / D_MODEL);
    const float var = tq * (1.0f / D_MODEL) - mu * mu;
    const float ri = rsqrtf(var + 1e-5f);
#pragma unroll
    for (int i = 0; i < 4; ++i) {
        const int idx = t + i * 256;
        const float y = (v[i] - mu) * ri * g[idx] + be[idx];
        if (out) out[(size_t)row * D_MODEL + idx] = y;
        if (outh) outh[(size_t)row * D_MODEL + idx] = (_Float16)y;
    }
}

// ---------------- host launch ----------------
extern "C" void kernel_launch(void* const* d_in, const int* in_sizes, int n_in,
                              void* d_out, int out_size, void* d_ws, size_t ws_size,
                              hipStream_t stream) {
    const float* x      = (const float*)d_in[0];
    const int*   seg    = (const int*)d_in[1];
    const float* qkv_w  = (const float*)d_in[2];
    const float* qkv_b  = (const float*)d_in[3];
    const float* out_w  = (const float*)d_in[4];
    const float* out_b  = (const float*)d_in[5];
    const float* ff_w1  = (const float*)d_in[6];
    const float* ff_b1  = (const float*)d_in[7];
    const float* ff_w2  = (const float*)d_in[8];
    const float* ff_b2  = (const float*)d_in[9];
    const float* ln1_g  = (const float*)d_in[10];
    const float* ln1_b  = (const float*)d_in[11];
    const float* ln2_g  = (const float*)d_in[12];
    const float* ln2_b  = (const float*)d_in[13];
    float* out = (float*)d_out;

    char* ws = (char*)d_ws;
    const size_t MB = 1024ull * 1024ull;
    _Float16* wt_qkv = (_Float16*)(ws + 0);         //  6 MB (3072x1024)
    _Float16* wt_out = (_Float16*)(ws + 6 * MB);    //  2 MB (1024x1024)
    _Float16* wt_ff1 = (_Float16*)(ws + 8 * MB);    //  8 MB (4096x1024)
    _Float16* wt_ff2 = (_Float16*)(ws + 16 * MB);   //  8 MB (1024x4096)
    _Float16* xh     = (_Float16*)(ws + 24 * MB);   // 16 MB
    _Float16* qh     = (_Float16*)(ws + 40 * MB);   // 16 MB
    _Float16* kh     = (_Float16*)(ws + 56 * MB);   // 16 MB
    _Float16* vt     = (_Float16*)(ws + 72 * MB);   // 16 MB (transposed V)
    _Float16* oh     = (_Float16*)(ws + 88 * MB);   // 16 MB
    float*    x1     = (float*)(ws + 104 * MB);     // 32 MB
    _Float16* x1h    = (_Float16*)(ws + 136 * MB);  // 16 MB
    float*    tmp    = (float*)(ws + 152 * MB);     // 32 MB
    _Float16* hb     = (_Float16*)(ws + 40 * MB);   // 64 MB, overlays q/k/v/o

    dim3 blk(256);
    // 0) weight f16 transposes + x f16
    wtrans_kernel<<<dim3(D_MODEL / 32, (3 * D_MODEL) / 32), blk, 0, stream>>>(
        qkv_w, wt_qkv, D_MODEL, 3 * D_MODEL);
    wtrans_kernel<<<dim3(D_MODEL / 32, D_MODEL / 32), blk, 0, stream>>>(
        out_w, wt_out, D_MODEL, D_MODEL);
    wtrans_kernel<<<dim3(D_MODEL / 32, D_FF / 32), blk, 0, stream>>>(
        ff_w1, wt_ff1, D_MODEL, D_FF);
    wtrans_kernel<<<dim3(D_FF / 32, D_MODEL / 32), blk, 0, stream>>>(
        ff_w2, wt_ff2, D_FF, D_MODEL);
    {
        const int n4 = M_ROWS * D_MODEL / 4;
        cvt16_kernel<<<dim3((n4 + 255) / 256), blk, 0, stream>>>(x, xh, n4);
    }
    // 1) QKV projection -> f16 q/k/v_t
    gemm_wmma_kernel<2><<<dim3((3 * D_MODEL) / 128, M_ROWS / 128), blk, 0, stream>>>(
        xh, wt_qkv, qkv_b, nullptr, nullptr, qh, kh, vt, M_ROWS, 3 * D_MODEL, D_MODEL);
    // 2) segment-masked flash attention (band-sparse) -> f16 o
    attn_kernel<<<dim3(SEQ / 128, BATCH * N_HEADS), blk, 0, stream>>>(qh, kh, vt, seg, oh);
    // 3) output projection -> f32 tmp
    gemm_wmma_kernel<0><<<dim3(D_MODEL / 128, M_ROWS / 128), blk, 0, stream>>>(
        oh, wt_out, out_b, tmp, nullptr, nullptr, nullptr, nullptr, M_ROWS, D_MODEL, D_MODEL);
    // 4) x1 = LN(x + attn_out), f32 + f16
    ln_kernel<<<dim3(M_ROWS), blk, 0, stream>>>(x, tmp, ln1_g, ln1_b, x1, x1h);
    // 5) h = gelu(x1 @ ff_w1 + b1) -> f16
    gemm_wmma_kernel<1><<<dim3(D_FF / 128, M_ROWS / 128), blk, 0, stream>>>(
        x1h, wt_ff1, ff_b1, nullptr, hb, nullptr, nullptr, nullptr, M_ROWS, D_FF, D_MODEL);
    // 6) ff = h @ ff_w2 + b2 -> f32 tmp
    gemm_wmma_kernel<0><<<dim3(D_MODEL / 128, M_ROWS / 128), blk, 0, stream>>>(
        hb, wt_ff2, ff_b2, tmp, nullptr, nullptr, nullptr, nullptr, M_ROWS, D_MODEL, D_FF);
    // 7) out = LN(x1 + ff)
    ln_kernel<<<dim3(M_ROWS), blk, 0, stream>>>(x1, tmp, ln2_g, ln2_b, out, nullptr);
}
